// STGCNBlock_90417651516131
// MI455X (gfx1250) — matrix operand order repaired and checked by reading, not compiled, not run on hardware
//
#include <hip/hip_runtime.h>
#include <hip/hip_bf16.h>

typedef __attribute__((ext_vector_type(16))) __bf16 v16bf;
typedef __attribute__((ext_vector_type(8)))  __bf16 bf8v;   // 16B
typedef __attribute__((ext_vector_type(8)))  float  v8f;

constexpr int NB = 8, NC = 64, NN = 1024, NT = 64, NE = 16384;
constexpr int BT   = NB * NT;     // 512
constexpr int FEAT = BT * NC;     // 32768 bf16 per node row
constexpr int TROW = 88;          // padded LDS row (176B: 44-dword lane stride -> no bank conflicts)

__device__ inline v8f wmma_bf16(v16bf a, v16bf b, v8f c) {
  return __builtin_amdgcn_wmma_f32_16x16x32_bf16(false, a, false, b, (short)0, c, false, false);
}
__device__ inline v16bf cat16(bf8v lo, bf8v hi) {
  return __builtin_shufflevector(lo, hi, 0,1,2,3,4,5,6,7,8,9,10,11,12,13,14,15);
}

// ---------------------------------------------------------------------------
// Kernel 1: weight prep. wT[k][co][ci] bf16 (per-tap GEMM A-matrices),
// gwbT[d][c] bf16 (transposed GCN weight), zero graph counters.
// ---------------------------------------------------------------------------
__global__ __launch_bounds__(256)
void prep_kernel(const float* __restrict__ w1, const float* __restrict__ w2,
                 const float* __restrict__ gw,
                 __bf16* __restrict__ wT1, __bf16* __restrict__ wT2,
                 __bf16* __restrict__ gwbT,
                 float* __restrict__ degW, int* __restrict__ cnt) {
  const int i = blockIdx.x * 256 + threadIdx.x;
  if (i < 9 * NC * NC) {
    const int k = i >> 12, co = (i >> 6) & 63, ci = i & 63;
    wT1[i] = (__bf16)w1[(co * 64 + ci) * 9 + k];
    wT2[i] = (__bf16)w2[(co * 64 + ci) * 9 + k];
  }
  if (i < NC * NC) {
    const int d = i >> 6, c = i & 63;
    gwbT[i] = (__bf16)gw[c * 64 + d];
  }
  if (i < NN) { degW[i] = 0.f; cnt[i] = 0; }
}

// ---------------------------------------------------------------------------
// Kernel 2/10: 1x9 temporal conv as 9 tap-GEMMs of 64x64x64 per (b,n).
// One block (128 thr = 4 waves) per (b,n); wave owns co rows [16w,16w+16).
// mode 0: input fp32 [B,C,N,T];   mode 1: input bf16 [N][BT][C]
// Output fp32 in [B][N][T][C] layout (channel-fastest).
// ---------------------------------------------------------------------------
__global__ __launch_bounds__(128)
void conv9_kernel(const float* __restrict__ xin_f32,
                  const __bf16* __restrict__ xin_bf16,
                  const __bf16* __restrict__ wT,     // [9][64co][64ci]
                  const float* __restrict__ bias,    // [64]
                  float* __restrict__ yout,          // [B][N][T][C] fp32
                  int mode) {
  __shared__ __bf16 xst[NT + 8][TROW];               // [t+4][ci], rows padded

  const int blk = blockIdx.x;
  const int n = blk & (NN - 1);
  const int b = blk >> 10;
  const int tid = threadIdx.x;

  // zero the +-4 halo rows (cols 0..63 used)
  for (int i = tid; i < 8 * 64; i += 128) {
    const int r = i >> 6, c = i & 63;
    xst[r < 4 ? r : 64 + r][c] = (__bf16)0.f;
  }
  if (mode == 0) {
    const int ci = tid >> 1, t0 = (tid & 1) * 32;
    const float* src = xin_f32 + (((size_t)b * NC + ci) * NN + n) * NT + t0;
    #pragma unroll
    for (int j = 0; j < 32; j += 4) {                // transpose scatter
      float4 v = *(const float4*)(src + j);
      xst[4 + t0 + j + 0][ci] = (__bf16)v.x;
      xst[4 + t0 + j + 1][ci] = (__bf16)v.y;
      xst[4 + t0 + j + 2][ci] = (__bf16)v.z;
      xst[4 + t0 + j + 3][ci] = (__bf16)v.w;
    }
  } else {
    const int t = tid >> 1, c0 = (tid & 1) * 32;     // tile already [t][c]
    const __bf16* src = xin_bf16 + ((size_t)n * BT + b * NT + t) * NC + c0;
    #pragma unroll
    for (int j = 0; j < 32; j += 8)
      *(bf8v*)&xst[4 + t][c0 + j] = *(const bf8v*)(src + j);
  }
  __syncthreads();

  const int wv = tid >> 5, lane = tid & 31;
  const int lm = lane & 15, lh = lane >> 4;
  const int co0 = wv * 16;
  v8f acc[4] = {};

  #pragma unroll
  for (int k = 0; k < 9; ++k) {
    const __bf16* wk = wT + k * (NC * NC) + (size_t)(co0 + lm) * NC;
    #pragma unroll
    for (int s = 0; s < 2; ++s) {
      // A fragment: rows m=co, K=ci block s*32..+31  (two contiguous 16B loads)
      const bf8v alo = *(const bf8v*)(wk + s * 32 + (lh << 3));
      const bf8v ahi = *(const bf8v*)(wk + s * 32 + 16 + (lh << 3));
      const v16bf a = cat16(alo, ahi);
      #pragma unroll
      for (int nt = 0; nt < 4; ++nt) {
        const int t = nt * 16 + lm;
        const __bf16* bp = &xst[t + k][s * 32 + (lh << 3)];  // two 16B loads: k 8h..+7, 16+8h..+7
        const v16bf bb = cat16(*(const bf8v*)bp, *(const bf8v*)(bp + 16));
        acc[nt] = wmma_bf16(a, bb, acc[nt]);
      }
    }
  }
  // epilogue: +bias, contiguous 8-channel stores per lane
  const float4 blo = *(const float4*)&bias[co0 + (lh << 3)];
  const float4 bhi = *(const float4*)&bias[co0 + (lh << 3) + 4];
  #pragma unroll
  for (int nt = 0; nt < 4; ++nt) {
    const int t = nt * 16 + lm;
    float* yp = yout + (((size_t)blk) * NT + t) * NC + co0 + (lh << 3);
    float4 lo = { acc[nt][0] + blo.x, acc[nt][1] + blo.y, acc[nt][2] + blo.z, acc[nt][3] + blo.w };
    float4 hi = { acc[nt][4] + bhi.x, acc[nt][5] + bhi.y, acc[nt][6] + bhi.z, acc[nt][7] + bhi.w };
    *(float4*)yp = lo;
    *(float4*)(yp + 4) = hi;
  }
}

// ---------------------------------------------------------------------------
// Kernel 3/11: BN statistics over [B][N][T][C] (c fastest).
// Block = 64K contiguous floats; thread owns a fixed 4-channel group.
// ---------------------------------------------------------------------------
__global__ __launch_bounds__(256)
void bn_stats_kernel(const float* __restrict__ y,
                     float* __restrict__ part_s, float* __restrict__ part_q) {
  const float* p = y + (size_t)blockIdx.x * 65536;
  const int tid = threadIdx.x;
  float4 s = {0.f, 0.f, 0.f, 0.f}, q = {0.f, 0.f, 0.f, 0.f};
  for (int i = tid * 4; i < 65536; i += 1024) {      // stride 1024 == 0 mod 64
    const float4 v = *(const float4*)(p + i);
    s.x += v.x; s.y += v.y; s.z += v.z; s.w += v.w;
    q.x += v.x * v.x; q.y += v.y * v.y; q.z += v.z * v.z; q.w += v.w * v.w;
  }
  __shared__ float4 sd_s[256], sd_q[256];
  sd_s[tid] = s; sd_q[tid] = q;
  __syncthreads();
  if (tid < 16) {                                    // group g covers channels 4g..4g+3
    float4 as = sd_s[tid], aq = sd_q[tid];
    for (int j = 1; j < 16; ++j) {
      const float4 bs = sd_s[tid + 16 * j], bq = sd_q[tid + 16 * j];
      as.x += bs.x; as.y += bs.y; as.z += bs.z; as.w += bs.w;
      aq.x += bq.x; aq.y += bq.y; aq.z += bq.z; aq.w += bq.w;
    }
    *(float4*)&part_s[blockIdx.x * 64 + tid * 4] = as;
    *(float4*)&part_q[blockIdx.x * 64 + tid * 4] = aq;
  }
}

__global__ void bn_finalize_kernel(const float* __restrict__ part_s,
                                   const float* __restrict__ part_q,
                                   const float* __restrict__ gamma, const float* __restrict__ beta,
                                   float* __restrict__ scale, float* __restrict__ shift) {
  const int c = threadIdx.x;                         // 64 threads
  float s = 0.f, q = 0.f;
  for (int blk = 0; blk < 512; ++blk) { s += part_s[blk * 64 + c]; q += part_q[blk * 64 + c]; }
  const float M = (float)NB * NN * NT;
  const float mean = s / M;
  const float var  = q / M - mean * mean;
  const float sc = gamma[c] * rsqrtf(var + 1e-5f);
  scale[c] = sc;
  shift[c] = beta[c] - mean * sc;
}

// ---------------------------------------------------------------------------
// Kernel 5: GCN linear. Per (b,n): D[d][t] = gw^T[d][c] x relu(bn(y1))[c][t].
// A = gwbT (global, contiguous), B = act[t][c] (LDS, contiguous).
// Output h bf16 in node-major [N][BT][C]; epilogue = 16B contiguous stores.
// ---------------------------------------------------------------------------
__global__ __launch_bounds__(128)
void gcn_lin_kernel(const float* __restrict__ y1,    // [B][N][T][C]
                    const float* __restrict__ scale, const float* __restrict__ shift,
                    const __bf16* __restrict__ gwbT, // [d][c]
                    __bf16* __restrict__ h) {        // [N][BT][C]
  __shared__ __bf16 act[NT][TROW];                   // [t][c] bn+relu'd

  const int blk = blockIdx.x;
  const int n = blk & (NN - 1);
  const int b = blk >> 10;
  const int tid = threadIdx.x;
  {
    const int t = tid >> 1, c0 = (tid & 1) * 32;
    const float* src = y1 + ((size_t)blk * NT + t) * NC + c0;
    #pragma unroll
    for (int jj = 0; jj < 32; jj += 8) {
      bf8v hv;
      #pragma unroll
      for (int u = 0; u < 8; u += 4) {
        const float4 v   = *(const float4*)(src + jj + u);
        const float4 scv = *(const float4*)(scale + c0 + jj + u);
        const float4 shv = *(const float4*)(shift + c0 + jj + u);
        hv[u + 0] = (__bf16)fmaxf(v.x * scv.x + shv.x, 0.f);
        hv[u + 1] = (__bf16)fmaxf(v.y * scv.y + shv.y, 0.f);
        hv[u + 2] = (__bf16)fmaxf(v.z * scv.z + shv.z, 0.f);
        hv[u + 3] = (__bf16)fmaxf(v.w * scv.w + shv.w, 0.f);
      }
      *(bf8v*)&act[t][c0 + jj] = hv;
    }
  }
  __syncthreads();

  const int wv = tid >> 5, lane = tid & 31;
  const int lm = lane & 15, lh = lane >> 4;
  const int d0w = wv * 16;
  v8f acc[4] = {};

  #pragma unroll
  for (int s = 0; s < 2; ++s) {                      // K = c = 64
    const __bf16* wr = gwbT + (size_t)(d0w + lm) * NC + s * 32;
    const v16bf a = cat16(*(const bf8v*)(wr + (lh << 3)),
                          *(const bf8v*)(wr + 16 + (lh << 3)));
    #pragma unroll
    for (int nt = 0; nt < 4; ++nt) {
      const __bf16* bp = &act[nt * 16 + lm][s * 32 + (lh << 3)];
      const v16bf bb = cat16(*(const bf8v*)bp, *(const bf8v*)(bp + 16));
      acc[nt] = wmma_bf16(a, bb, acc[nt]);
    }
  }
  #pragma unroll
  for (int nt = 0; nt < 4; ++nt) {                   // M=d rows, N=t lanes
    const int t = nt * 16 + lm;
    const int d0 = d0w + (lh << 3);
    bf8v hv;
    #pragma unroll
    for (int r = 0; r < 8; ++r) hv[r] = (__bf16)acc[nt][r];
    *(bf8v*)&h[((size_t)n * BT + b * NT + t) * NC + d0] = hv;
  }
}

// ---------------------------------------------------------------------------
// Graph prep: degree, scan -> CSR rowptr, fill (counting sort keyed on col)
// ---------------------------------------------------------------------------
__global__ __launch_bounds__(256)
void graph_deg_kernel(const int* __restrict__ ei, const float* __restrict__ ew,
                      float* __restrict__ degW, int* __restrict__ cnt) {
  const int i = blockIdx.x * 256 + threadIdx.x;
  if (i < NE) {
    const int col = ei[NE + i];
    atomicAdd(&degW[col], ew[i]);
    atomicAdd(&cnt[col], 1);
  } else if (i < NE + NN) {
    const int nn = i - NE;                           // self loop, weight 1
    atomicAdd(&degW[nn], 1.0f);
    atomicAdd(&cnt[nn], 1);
  }
}

__global__ __launch_bounds__(1024)
void graph_scan_kernel(const float* __restrict__ degW, const int* __restrict__ cnt,
                       float* __restrict__ dinv, int* __restrict__ rowptr,
                       int* __restrict__ fillptr) {
  const int n = threadIdx.x;
  const float d = degW[n];
  dinv[n] = d > 0.f ? rsqrtf(d) : 0.f;
  __syncthreads();
  if (n == 0) {
    int acc = 0;
    for (int i = 0; i < NN; ++i) { rowptr[i] = acc; fillptr[i] = acc; acc += cnt[i]; }
    rowptr[NN] = acc;                                // == NE + NN
  }
}

__global__ __launch_bounds__(256)
void graph_fill_kernel(const int* __restrict__ ei, const float* __restrict__ ew,
                       const float* __restrict__ dinv, int* __restrict__ fillptr,
                       int* __restrict__ srcn, float* __restrict__ val) {
  const int i = blockIdx.x * 256 + threadIdx.x;
  if (i < NE) {
    const int row = ei[i], col = ei[NE + i];
    const int pos = atomicAdd(&fillptr[col], 1);
    srcn[pos] = row;
    val[pos]  = dinv[row] * ew[i] * dinv[col];
  } else if (i < NE + NN) {
    const int nn = i - NE;
    const int pos = atomicAdd(&fillptr[nn], 1);
    srcn[pos] = nn;
    val[pos]  = dinv[nn] * dinv[nn];
  }
}

// ---------------------------------------------------------------------------
// Kernel 9: SpMM aggregate + bias + ReLU. h = 67MB bf16 -> L2-resident (192MB).
// ---------------------------------------------------------------------------
__global__ __launch_bounds__(256)
void spmm_kernel(const __bf16* __restrict__ h,       // [N][FEAT]
                 const int* __restrict__ rowptr, const int* __restrict__ srcn,
                 const float* __restrict__ val, const float* __restrict__ gb,
                 __bf16* __restrict__ x2) {          // [N][FEAT]
  const int n = blockIdx.x >> 3;
  const int base = (blockIdx.x & 7) * 4096 + threadIdx.x * 16;
  float acc[16] = {};
  const int p0 = rowptr[n], p1 = rowptr[n + 1];
  for (int p = p0; p < p1; ++p) {
    const int s = srcn[p];
    const float v = val[p];
    if (p + 1 < p1)
      __builtin_prefetch(h + (size_t)srcn[p + 1] * FEAT + base, 0, 1);
    const v16bf hv = *(const v16bf*)(h + (size_t)s * FEAT + base);
    #pragma unroll
    for (int j = 0; j < 16; ++j) acc[j] += v * (float)hv[j];
  }
  const int c0 = base & (NC - 1);
  __bf16* op = x2 + (size_t)n * FEAT + base;
  #pragma unroll
  for (int j = 0; j < 16; ++j) {
    const float r = acc[j] + gb[c0 + j];
    op[j] = (__bf16)(r > 0.f ? r : 0.f);
  }
}

// ---------------------------------------------------------------------------
// Kernel 13: out = relu(bn2(y2)) + x.  y2 is [B][N][T][C]; out is [B][C][N][T].
// LDS tile transpose per (b,n): 64t x 64c.
// ---------------------------------------------------------------------------
__global__ __launch_bounds__(256)
void final_kernel(const float* __restrict__ y2,      // [B][N][T][C]
                  const float* __restrict__ x,       // [B][C][N][T]
                  const float* __restrict__ scale, const float* __restrict__ shift,
                  float* __restrict__ out) {         // [B][C][N][T]
  __shared__ float tile[64][65];                     // [t][c], padded
  const int blk = blockIdx.x;
  const int n = blk & (NN - 1);
  const int b = blk >> 10;
  const int tid = threadIdx.x;

  const float* yp = y2 + (size_t)blk * 4096;
  for (int i = tid * 4; i < 4096; i += 1024) {
    const float4 v = *(const float4*)(yp + i);
    const int t = i >> 6, c = i & 63;
    tile[t][c + 0] = v.x; tile[t][c + 1] = v.y;
    tile[t][c + 2] = v.z; tile[t][c + 3] = v.w;
  }
  __syncthreads();

  const int c = tid >> 2, t0 = (tid & 3) * 16;
  const float sc = scale[c], sh = shift[c];
  const size_t base = (((size_t)b * NC + c) * NN + n) * NT + t0;
  #pragma unroll
  for (int j = 0; j < 16; j += 4) {
    const float4 r = *(const float4*)(x + base + j);
    float4 o;
    o.x = fmaxf(tile[t0 + j + 0][c] * sc + sh, 0.f) + r.x;
    o.y = fmaxf(tile[t0 + j + 1][c] * sc + sh, 0.f) + r.y;
    o.z = fmaxf(tile[t0 + j + 2][c] * sc + sh, 0.f) + r.z;
    o.w = fmaxf(tile[t0 + j + 3][c] * sc + sh, 0.f) + r.w;
    *(float4*)(out + base + j) = o;
  }
}

// ---------------------------------------------------------------------------
extern "C" void kernel_launch(void* const* d_in, const int* in_sizes, int n_in,
                              void* d_out, int out_size, void* d_ws, size_t ws_size,
                              hipStream_t stream) {
  const float* x   = (const float*)d_in[0];
  const int*   ei  = (const int*)  d_in[1];
  const float* ew  = (const float*)d_in[2];
  const float* w1  = (const float*)d_in[3];
  const float* b1  = (const float*)d_in[4];
  const float* g1  = (const float*)d_in[5];
  const float* be1 = (const float*)d_in[6];
  const float* gw  = (const float*)d_in[7];
  const float* gb  = (const float*)d_in[8];
  const float* w2  = (const float*)d_in[9];
  const float* b2  = (const float*)d_in[10];
  const float* g2  = (const float*)d_in[11];
  const float* be2 = (const float*)d_in[12];
  float* out = (float*)d_out;

  char* p = (char*)d_ws;
  auto alloc = [&](size_t bytes) -> char* {
    char* r = p; p += (bytes + 255) & ~(size_t)255; return r;
  };
  const size_t tensor_elems = (size_t)NB * NC * NN * NT;       // 33.5M
  float*  y1    = (float*) alloc(tensor_elems * 4);            // [B][N][T][C]
  float*  y2    = (float*) alloc(tensor_elems * 4);            // [B][N][T][C]
  __bf16* h     = (__bf16*)alloc((size_t)NN * FEAT * 2);
  __bf16* x2    = (__bf16*)alloc((size_t)NN * FEAT * 2);
  __bf16* wT1   = (__bf16*)alloc(9 * NC * NC * 2);
  __bf16* wT2   = (__bf16*)alloc(9 * NC * NC * 2);
  __bf16* gwbT  = (__bf16*)alloc(NC * NC * 2);
  float*  psum1 = (float*) alloc(512 * 64 * 4);
  float*  pqsum1= (float*) alloc(512 * 64 * 4);
  float*  psum2 = (float*) alloc(512 * 64 * 4);
  float*  pqsum2= (float*) alloc(512 * 64 * 4);
  float*  sc1   = (float*) alloc(NC * 4);
  float*  sh1   = (float*) alloc(NC * 4);
  float*  sc2   = (float*) alloc(NC * 4);
  float*  sh2   = (float*) alloc(NC * 4);
  float*  degW  = (float*) alloc(NN * 4);
  int*    cnt   = (int*)   alloc(NN * 4);
  int*    rowptr= (int*)   alloc((NN + 1) * 4);
  int*    fillp = (int*)   alloc(NN * 4);
  float*  dinv  = (float*) alloc(NN * 4);
  int*    srcn  = (int*)   alloc((NE + NN) * 4);
  float*  val   = (float*) alloc((NE + NN) * 4);

  const int gemm_blocks  = NB * NN;                            // 8192
  const int graph_blocks = (NE + NN + 255) / 256;              // 68

  prep_kernel<<<(9 * NC * NC + 255) / 256, 256, 0, stream>>>(w1, w2, gw, wT1, wT2, gwbT, degW, cnt);

  conv9_kernel<<<gemm_blocks, 128, 0, stream>>>(x, x2, wT1, b1, y1, 0);
  bn_stats_kernel<<<512, 256, 0, stream>>>(y1, psum1, pqsum1);
  bn_finalize_kernel<<<1, NC, 0, stream>>>(psum1, pqsum1, g1, be1, sc1, sh1);

  gcn_lin_kernel<<<gemm_blocks, 128, 0, stream>>>(y1, sc1, sh1, gwbT, h);

  graph_deg_kernel<<<graph_blocks, 256, 0, stream>>>(ei, ew, degW, cnt);
  graph_scan_kernel<<<1, 1024, 0, stream>>>(degW, cnt, dinv, rowptr, fillp);
  graph_fill_kernel<<<graph_blocks, 256, 0, stream>>>(ei, ew, dinv, fillp, srcn, val);
  spmm_kernel<<<NN * 8, 256, 0, stream>>>(h, rowptr, srcn, val, gb, x2);

  conv9_kernel<<<gemm_blocks, 128, 0, stream>>>(x, x2, wT2, b2, y2, 1);
  bn_stats_kernel<<<512, 256, 0, stream>>>(y2, psum2, pqsum2);
  bn_finalize_kernel<<<1, NC, 0, stream>>>(psum2, pqsum2, g2, be2, sc2, sh2);

  final_kernel<<<gemm_blocks, 256, 0, stream>>>(y2, x, sc2, sh2, out);
}
